// ConfidenceProtoSegHead_47399259078720
// MI455X (gfx1250) — compile-verified, matrix operand order checked by reading, not run on hardware
//
#include <hip/hip_runtime.h>
#include <hip/hip_bf16.h>

// ---------------------------------------------------------------------------
// ConfidenceProtoSegHead for MI455X (gfx1250, wave32, WMMA).
//
// sim = xf[131072,256] @ protosT[256,190]  (bf16 WMMA, f32 accum)
// out_seg = layernorm(max over 10 protos per class)  -> [4,19,128,256]
//
// Roofline: 134 MB in + 10 MB out @ 23.3 TB/s => ~6.2 us floor. 12.7 GFLOP in
// the bf16 WMMA path is sub-floor compute; fp32 WMMA (16x16x4) would not be.
// ---------------------------------------------------------------------------

typedef __bf16 v16bf __attribute__((ext_vector_type(16)));
typedef __bf16 v2bf  __attribute__((ext_vector_type(2)));
typedef float  v8f   __attribute__((ext_vector_type(8)));

#define CLS      19
#define NPROTO   10
#define NCOL     190          // 19*10
#define NTILES   12           // N padded to 192 = 12 tiles of 16
#define KDIM     256
#define KTILES   8            // 256 / 32
#define Hh       128
#define Ww       256
#define HW       (Hh * Ww)    // 32768
#define CHW      (KDIM * HW)
#define NPIX     (4 * HW)     // 131072
#define LDSTRIDE 193          // 193 % 64 == 1 -> bank-conflict-free rows
#define PIX_PER_WG 64         // 4 waves * 16 pixels
#define BSWZ_WORDS (KTILES * NTILES * 32 * 8)   // 24576 dwords = 96 KB in d_ws

__device__ __forceinline__ unsigned int pack_bf16(float a, float b) {
  // fptrunc f32 -> bf16 (RNE); backend picks the native convert if available.
  v2bf p;
  p.x = (__bf16)a;
  p.y = (__bf16)b;
  return __builtin_bit_cast(unsigned int, p);
}

union Frag {
  unsigned int u[8];
  uint4        q[2];
  v16bf        bf;
};

// K index held by (lane-half, vgpr) for the 16-bit 16x32 A fragment
// (ISA 7.12.2); B staging mirrors this with M<->N swapped.
__device__ __forceinline__ int frag_kbase(int v, int half) {
  return (v < 4) ? (2 * v + half * 8) : (16 + 2 * (v - 4) + half * 8);
}

// ---------------------------------------------------------------------------
// Kernel 1: L2-normalize prototypes over feature dim, convert to bf16, and
// scatter into the per-lane WMMA B-fragment layout in d_ws.
// word index = ((kt*NTILES + nt)*32 + lane)*8 + v
// ---------------------------------------------------------------------------
__global__ __launch_bounds__(256) void proto_prep_kernel(
    const float* __restrict__ proto, unsigned int* __restrict__ Bswz) {
  const int gid = blockIdx.x * blockDim.x + threadIdx.x;
  if (gid >= BSWZ_WORDS) return;
  const int v    = gid & 7;
  const int lane = (gid >> 3) & 31;
  const int nt   = (gid >> 8) % NTILES;
  const int kt   = (gid >> 8) / NTILES;
  const int half = lane >> 4;
  const int n    = nt * 16 + (lane & 15);      // column: class*10 + proto
  unsigned int word = 0u;                      // pad columns 190,191 -> 0
  if (n < NCOL) {
    const float* row = proto + (size_t)n * KDIM;
    float ssq = 0.0f;
    for (int i = 0; i < KDIM; ++i) { float t = row[i]; ssq += t * t; }
    const float inv = 1.0f / (sqrtf(ssq) + 1e-12f);
    const int c = kt * 32 + frag_kbase(v, half);
    word = pack_bf16(row[c] * inv, row[c + 1] * inv);
  }
  Bswz[gid] = word;
}

// ---------------------------------------------------------------------------
// Kernel 2: GEMM (bf16 WMMA) + per-class max + LayerNorm.
// 4 waves / block, one 16-pixel M-tile per wave, all 12 N-tiles per wave.
// ---------------------------------------------------------------------------
__global__ __launch_bounds__(128, 1) void proto_seg_main_kernel(
    const float* __restrict__ x, const unsigned int* __restrict__ Bswz,
    const float* __restrict__ gamma, const float* __restrict__ beta,
    float* __restrict__ out) {
  __shared__ float simbuf[PIX_PER_WG * LDSTRIDE];   // ~48.3 KB

  const int tid  = threadIdx.x;
  const int wave = tid >> 5;
  const int lane = tid & 31;
  const int half = lane >> 4;
  const int m    = lane & 15;

  // this wave's 16 consecutive pixels (tile is W-row aligned: 16 | p0, 256=W)
  const int p0 = blockIdx.x * PIX_PER_WG + wave * 16;
  const int bi = p0 >> 15;               // / (H*W)
  const int hi = (p0 >> 8) & (Hh - 1);   // / W % H
  const int wb = p0 & (Ww - 1);          // % W
  const float* xbase = x + (size_t)bi * CHW + (size_t)hi * Ww + wb + m;

  v8f acc[NTILES] = {};                  // 96 VGPRs of f32 accumulators

  for (int kt = 0; kt < KTILES; ++kt) {
    if (kt + 1 < KTILES) {
      // gfx1250 global_prefetch_b8 on next K-tile's channel plane
      __builtin_prefetch(xbase + (size_t)((kt + 1) * 32) * HW, 0, 1);
    }
    // A fragment: 16 pixels x 32 channels, fp32 loads (coalesced across the
    // 16-pixel half-wave) -> packed bf16 in registers.
    Frag A;
#pragma unroll
    for (int v = 0; v < 8; ++v) {
      const int c  = kt * 32 + frag_kbase(v, half);
      const float f0 = xbase[(size_t)c * HW];
      const float f1 = xbase[(size_t)(c + 1) * HW];
      A.u[v] = pack_bf16(f0, f1);
    }
    // B fragments: pre-swizzled, 2 x b128 per lane per tile (L2-resident).
    const unsigned int* bbase = Bswz + ((size_t)(kt * NTILES) * 32 + lane) * 8;
#pragma unroll
    for (int nt = 0; nt < NTILES; ++nt) {
      Frag B;
      const uint4* bq = (const uint4*)(bbase + (size_t)nt * (32 * 8));
      B.q[0] = bq[0];
      B.q[1] = bq[1];
      acc[nt] = __builtin_amdgcn_wmma_f32_16x16x32_bf16(
          /*neg_a=*/false, A.bf, /*neg_b=*/false, B.bf,
          /*c_mod=*/(short)0, acc[nt], /*reuse_a=*/false, /*reuse_b=*/false);
    }
  }

  // Spill sim tiles to LDS (C/D layout: vgpr r -> row r + 8*half, lane&15 -> N)
  const int prow = wave * 16;
#pragma unroll
  for (int nt = 0; nt < NTILES; ++nt) {
#pragma unroll
    for (int v = 0; v < 8; ++v) {
      simbuf[(prow + half * 8 + v) * LDSTRIDE + nt * 16 + m] = acc[nt][v];
    }
  }
  __syncthreads();

  // Phase 2: per-pixel max over 10 protos per class, then LayerNorm over 19.
  if (tid < PIX_PER_WG) {
    const float* row = &simbuf[tid * LDSTRIDE];
    float mx[CLS];
    float s = 0.0f, ssq = 0.0f;
#pragma unroll
    for (int k = 0; k < CLS; ++k) {
      float v0 = row[k * NPROTO];
#pragma unroll
      for (int j = 1; j < NPROTO; ++j) v0 = fmaxf(v0, row[k * NPROTO + j]);
      mx[k] = v0;
      s += v0;
      ssq += v0 * v0;
    }
    const float mean = s * (1.0f / (float)CLS);
    const float var  = fmaxf(ssq * (1.0f / (float)CLS) - mean * mean, 0.0f);
    const float rstd = rsqrtf(var + 1e-5f);

    const int P    = blockIdx.x * PIX_PER_WG + tid;
    const int obi  = P >> 15;
    const int orem = P & (HW - 1);
    float* obase = out + (size_t)obi * CLS * HW + orem;
#pragma unroll
    for (int k = 0; k < CLS; ++k) {
      obase[(size_t)k * HW] = (mx[k] - mean) * rstd * gamma[k] + beta[k];
    }
  }
}

// ---------------------------------------------------------------------------
extern "C" void kernel_launch(void* const* d_in, const int* in_sizes, int n_in,
                              void* d_out, int out_size, void* d_ws,
                              size_t ws_size, hipStream_t stream) {
  const float* x     = (const float*)d_in[0];  // [4,256,128,256]
  const float* proto = (const float*)d_in[1];  // [19,10,256]
  const float* gamma = (const float*)d_in[2];  // [19]
  const float* beta  = (const float*)d_in[3];  // [19]
  float* out = (float*)d_out;                  // [4,19,128,256]
  unsigned int* Bswz = (unsigned int*)d_ws;    // 96 KB of swizzled bf16 protos

  (void)in_sizes; (void)n_in; (void)out_size; (void)ws_size;

  // Stage normalized bf16 prototypes in WMMA B-fragment order.
  proto_prep_kernel<<<(BSWZ_WORDS + 255) / 256, 256, 0, stream>>>(proto, Bswz);

  // Main fused GEMM + max + LayerNorm.
  proto_seg_main_kernel<<<NPIX / PIX_PER_WG, 128, 0, stream>>>(
      x, Bswz, gamma, beta, out);
}
